// FraudDetectionHybrid_65481071408210
// MI455X (gfx1250) — matrix operand order repaired and checked by reading, not compiled, not run on hardware
//
#include <hip/hip_runtime.h>
#include <hip/hip_bf16.h>

typedef __bf16 bf16_t;
typedef bf16_t v16bf __attribute__((ext_vector_type(16)));
typedef bf16_t v8bf  __attribute__((ext_vector_type(8)));
typedef float  v8f   __attribute__((ext_vector_type(8)));

static constexpr int HDIM = 2048;   // hidden dim
static constexpr int BM   = 32;     // rows per block (2 WMMA M-tiles)

// ---------------------------------------------------------------------------
// fast tanh via v_exp_f32 / v_rcp_f32: tanh(x) = sign(x)*(1 - 2/(e^{2|x|}+1))
// ---------------------------------------------------------------------------
__device__ __forceinline__ float fast_tanh(float x) {
  float ax = __builtin_fabsf(x);
  float e  = __builtin_amdgcn_exp2f(ax * 2.8853900817779268f);  // 2*log2(e)
  float t  = 1.0f - 2.0f * __builtin_amdgcn_rcpf(e + 1.0f);
  return __builtin_copysignf(t, x);
}

__device__ __forceinline__ v8f wmma_bf16(v16bf a, v16bf b, v8f c) {
  // D = A(16x32 bf16) * B(32x16 bf16) + C(16x16 f32)
  return __builtin_amdgcn_wmma_f32_16x16x32_bf16(
      /*neg_a=*/false, a, /*neg_b=*/false, b,
      /*c_mod=*/(short)0, c, /*reuse_a=*/false, /*reuse_b=*/false);
}

// ---------------------------------------------------------------------------
// Pack W2 (f32, row-major [K][N]) -> w2t (bf16, [N][K]) so B-fragments are
// contiguous 32B per lane.  Tiled transpose through LDS.
// ---------------------------------------------------------------------------
__global__ __launch_bounds__(256) void pack_w2_kernel(const float* __restrict__ W2,
                                                      bf16_t* __restrict__ w2t) {
  __shared__ float tile[32][33];
  const int kb = blockIdx.x * 32;
  const int nb = blockIdx.y * 32;
  const int tx = threadIdx.x & 31;
  const int ty = threadIdx.x >> 5;  // 0..7
#pragma unroll
  for (int i = ty; i < 32; i += 8)
    tile[i][tx] = W2[(kb + i) * HDIM + nb + tx];
  __syncthreads();
#pragma unroll
  for (int i = ty; i < 32; i += 8)
    w2t[(nb + i) * HDIM + kb + tx] = (bf16_t)tile[tx][i];
}

// ---------------------------------------------------------------------------
// Fused: qfeat -> L1 (K=6, VALU) -> LDS bf16 tile -> L2 GEMM (WMMA bf16) ->
// tanh -> dot Wc (fused classifier) ; plus the tiny regressor branch.
// ---------------------------------------------------------------------------
__global__ __launch_bounds__(256)
__attribute__((amdgpu_waves_per_eu(4)))
void fused_kernel(
    const float* __restrict__ x,
    const float* __restrict__ W1,  const float* __restrict__ b1,
    const float* __restrict__ b2,
    const float* __restrict__ Wc,  const float* __restrict__ bc,
    const float* __restrict__ Wr1, const float* __restrict__ br1,
    const float* __restrict__ Wr2, const float* __restrict__ br2,
    const float* __restrict__ Wr3, const float* __restrict__ br3,
    const bf16_t* __restrict__ w2t,
    float* __restrict__ out, int Btot) {

  __shared__ bf16_t h1[BM][HDIM + 8];   // padded: conflict-free A-frag loads
  __shared__ float  cfeat[BM][4];       // x0, x1, cos x0, cos x0 * cos x1
  __shared__ float  logits_lds[BM];

  const int tid     = threadIdx.x;
  const int rowBase = blockIdx.x * BM;

  // ---- feature map (concat = [x0, x1, c0, c0*c1, 1, 1]) ----
  if (tid < BM) {
    float x0 = x[(rowBase + tid) * 2 + 0];
    float x1 = x[(rowBase + tid) * 2 + 1];
    float c0 = cosf(x0);
    float c1 = cosf(x1);
    cfeat[tid][0] = x0;
    cfeat[tid][1] = x1;
    cfeat[tid][2] = c0;
    cfeat[tid][3] = c0 * c1;
    logits_lds[tid] = 0.0f;
  }
  __syncthreads();

  // ---- layer 1: h1 = tanh(concat @ W1 + b1), K = 6 (two inputs are 1.0) ----
  for (int idx = tid; idx < BM * HDIM; idx += 256) {
    const int m = idx >> 11;
    const int n = idx & (HDIM - 1);
    float v = b1[n] + W1[4 * HDIM + n] + W1[5 * HDIM + n];
    v += cfeat[m][0] * W1[0 * HDIM + n];
    v += cfeat[m][1] * W1[1 * HDIM + n];
    v += cfeat[m][2] * W1[2 * HDIM + n];
    v += cfeat[m][3] * W1[3 * HDIM + n];
    h1[m][n] = (bf16_t)fast_tanh(v);
  }
  __syncthreads();

  // ---- layer 2 + classifier: logits = tanh(h1 @ W2 + b2) @ Wc + bc ----
  const int lane  = tid & 31;
  const int wave  = tid >> 5;          // 0..7, each owns 16 N-tiles
  const int nlo   = lane & 15;
  const int aoff  = (lane >> 4) << 4;  // A k-offset: 0 or 16 (contiguous, matches B)
  const int boff  = (lane >> 4) << 4;  // B k-offset: 0 or 16 (contiguous, matches A)

  const bf16_t* aRow0 = &h1[nlo][aoff];       // M-tile 0: rows 0..15
  const bf16_t* aRow1 = &h1[16 + nlo][aoff];  // M-tile 1: rows 16..31

  float p[2][8];
#pragma unroll
  for (int mt = 0; mt < 2; ++mt)
#pragma unroll
    for (int r = 0; r < 8; ++r) p[mt][r] = 0.0f;

  const v8f zacc = {0.f, 0.f, 0.f, 0.f, 0.f, 0.f, 0.f, 0.f};

#pragma unroll 1
  for (int ng = 0; ng < 4; ++ng) {            // 4 groups of 4 N-tiles
    const int ntb = wave * 16 + ng * 4;
    // group base: N-tile j is a constant +j*16*HDIM element offset (imm IOFFSET)
    const bf16_t* pg = w2t + (ntb * 16 + nlo) * HDIM + boff;

    v8f acc[4][2];
    float b2n[4], wcn[4];
#pragma unroll
    for (int j = 0; j < 4; ++j) {
      const int N = (ntb + j) * 16 + nlo;     // this lane's output column
      b2n[j]  = b2[N];
      wcn[j]  = Wc[N];
      acc[j][0] = zacc;
      acc[j][1] = zacc;
    }

#pragma unroll 1
    for (int k0 = 0; k0 < HDIM; k0 += 32) {   // 64 K-steps, no unroll: no spills
      // A fragments (shared across the 4 N-tiles of the group)
      v8bf a0lo = *(const v8bf*)(aRow0 + k0);
      v8bf a0hi = *(const v8bf*)(aRow0 + k0 + 8);
      v8bf a1lo = *(const v8bf*)(aRow1 + k0);
      v8bf a1hi = *(const v8bf*)(aRow1 + k0 + 8);
      v16bf a0 = __builtin_shufflevector(a0lo, a0hi, 0, 1, 2, 3, 4, 5, 6, 7,
                                         8, 9, 10, 11, 12, 13, 14, 15);
      v16bf a1 = __builtin_shufflevector(a1lo, a1hi, 0, 1, 2, 3, 4, 5, 6, 7,
                                         8, 9, 10, 11, 12, 13, 14, 15);
      const bf16_t* pk = pg + k0;
#pragma unroll
      for (int j = 0; j < 4; ++j) {
        v16bf bb = *(const v16bf*)(pk + j * (16 * HDIM));  // 32B contiguous
        acc[j][0] = wmma_bf16(a0, bb, acc[j][0]);
        acc[j][1] = wmma_bf16(a1, bb, acc[j][1]);
      }
    }

    // epilogue: tanh + classifier dot, accumulated in registers
#pragma unroll
    for (int j = 0; j < 4; ++j)
#pragma unroll
      for (int r = 0; r < 8; ++r) {
        p[0][r] += fast_tanh(acc[j][0][r] + b2n[j]) * wcn[j];
        p[1][r] += fast_tanh(acc[j][1][r] + b2n[j]) * wcn[j];
      }
  }

  // C/D layout: VGPR r -> M = r (lanes 0-15) or 8+r (lanes 16-31)
  const int mhalf = (lane >> 4) << 3;
#pragma unroll
  for (int r = 0; r < 8; ++r) {
    atomicAdd(&logits_lds[0  + mhalf + r], p[0][r]);
    atomicAdd(&logits_lds[16 + mhalf + r], p[1][r]);
  }
  __syncthreads();

  // ---- write logits + tiny regressor branch ----
  if (tid < BM) {
    const int row = rowBase + tid;
    out[row] = logits_lds[tid] + bc[0];

    const float x0 = cfeat[tid][0], x1 = cfeat[tid][1];
    float r1[8];
#pragma unroll
    for (int j = 0; j < 8; ++j)
      r1[j] = fast_tanh(br1[j] + x0 * Wr1[0 * 8 + j] + x1 * Wr1[1 * 8 + j]);
    float r2[4];
#pragma unroll
    for (int k = 0; k < 4; ++k) {
      float s = br2[k];
#pragma unroll
      for (int j = 0; j < 8; ++j) s += r1[j] * Wr2[j * 4 + k];
      r2[k] = fast_tanh(s);
    }
    float risk = br3[0];
#pragma unroll
    for (int k = 0; k < 4; ++k) risk += r2[k] * Wr3[k];
    out[Btot + row] = risk;
  }
}

// ---------------------------------------------------------------------------
extern "C" void kernel_launch(void* const* d_in, const int* in_sizes, int n_in,
                              void* d_out, int out_size, void* d_ws, size_t ws_size,
                              hipStream_t stream) {
  const float* x   = (const float*)d_in[0];
  const float* W1  = (const float*)d_in[1];
  const float* b1  = (const float*)d_in[2];
  const float* W2  = (const float*)d_in[3];
  const float* b2  = (const float*)d_in[4];
  const float* Wc  = (const float*)d_in[5];
  const float* bc  = (const float*)d_in[6];
  const float* Wr1 = (const float*)d_in[7];
  const float* br1 = (const float*)d_in[8];
  const float* Wr2 = (const float*)d_in[9];
  const float* br2 = (const float*)d_in[10];
  const float* Wr3 = (const float*)d_in[11];
  const float* br3 = (const float*)d_in[12];
  float*  out = (float*)d_out;
  bf16_t* w2t = (bf16_t*)d_ws;           // needs HDIM*HDIM*2 = 8 MB scratch

  const int Btot = in_sizes[0] / 2;      // B = 32768

  dim3 gp(HDIM / 32, HDIM / 32);
  pack_w2_kernel<<<gp, 256, 0, stream>>>(W2, w2t);

  fused_kernel<<<Btot / BM, 256, 0, stream>>>(
      x, W1, b1, b2, Wc, bc, Wr1, br1, Wr2, br2, Wr3, br3, w2t, out, Btot);
}
